// GCNModel_9156870275646
// MI455X (gfx1250) — compile-verified, matrix operand order checked by reading, not run on hardware
//
#include <hip/hip_runtime.h>

// ---------------------------------------------------------------------------
// GCN (3x GCNConv) for MI455X / gfx1250, wave32 + WMMA bf16.
// N=100000 nodes, E=3200000 edges, dims 384 -> 128 -> 128 -> 5.
// GEMM: W repacked once into per-lane WMMA B-fragment layout (bf16, global,
// L2-resident) -> GEMM has no LDS, no barriers; pure vmem + wmma.
// ---------------------------------------------------------------------------

typedef __attribute__((ext_vector_type(16))) __bf16 v16bf;
typedef __attribute__((ext_vector_type(8)))  float  v8f;

#define NNODES 100000
#define NEDGES 3200000LL

// ------------------------- degree / norm kernels ---------------------------

__global__ void kInitDeg(float* __restrict__ deg, int n) {
    int i = blockIdx.x * blockDim.x + threadIdx.x;
    if (i < n) deg[i] = 1.0f;   // self-loop contribution
}

__global__ void kDegEdges(const long long* __restrict__ ei, float* __restrict__ deg,
                          long long E) {
    long long i = (long long)blockIdx.x * blockDim.x + threadIdx.x;
    if (i < E) {
        int d = (int)ei[E + i];   // dst row
        __hip_atomic_fetch_add(&deg[d], 1.0f, __ATOMIC_RELAXED, __HIP_MEMORY_SCOPE_AGENT);
    }
}

__global__ void kRsqrt(float* __restrict__ deg, int n) {
    int i = blockIdx.x * blockDim.x + threadIdx.x;
    if (i < n) {
        float v = deg[i];
        deg[i] = (v > 0.0f) ? rsqrtf(v) : 0.0f;
    }
}

// ------------------- W -> WMMA B-fragment repack (bf16) --------------------
// Fragment layout consumed by the GEMM:
//   Wfrag[ ((kchunk*NT + t)*32 + lane)*16 + j ]
// where for lane = half*16 + n (n = col within 16-wide tile):
//   j in [0,8):  K = kchunk*32 + half*8 + j
//   j in [8,16): K = kchunk*32 + 16 + half*8 + (j-8)
// and col = t*16 + n  (zero-padded past K_out_real).
template <int NT>
__global__ void kPackW(const float* __restrict__ W, __bf16* __restrict__ Wfrag,
                       int K_in, int K_out_real) {
    int idx = blockIdx.x * blockDim.x + threadIdx.x;
    int total = (K_in >> 5) * NT * 512;      // 512 = 32 lanes * 16 elems
    if (idx >= total) return;
    int j     = idx & 15;
    int lane  = (idx >> 4) & 31;
    int rest  = idx >> 9;                    // = kchunk*NT + t
    int t     = rest % NT;
    int chunk = rest / NT;
    int half  = lane >> 4;
    int n     = lane & 15;
    int k     = chunk * 32 + half * 8 + (j & 7) + ((j >> 3) << 4);
    int c     = t * 16 + n;
    float v   = (c < K_out_real) ? W[k * K_out_real + c] : 0.0f;
    Wfrag[idx] = (__bf16)v;
}

// ----------------------------- WMMA GEMM -----------------------------------
// C[N x (NT*16)] = X[N x KIN] @ W  (W pre-packed bf16 fragments in global).
// 320 threads = 10 waves per block; wave w owns row-tile blockIdx.x*10 + w.
// 6250 row tiles -> grid 625, exact fit, EXEC all 1s, no LDS, no barriers.
template <int NT, int KIN>
__global__ __launch_bounds__(320)
void gemm_wmma_bf16(const float* __restrict__ X, const __bf16* __restrict__ Wfrag,
                    float* __restrict__ C) {
    constexpr int KOUT    = NT * 16;
    constexpr int KCHUNKS = KIN / 32;

    const int tid  = threadIdx.x;
    const int wave = tid >> 5;
    const int lane = tid & 31;
    const int half = lane >> 4;     // 0: lanes 0-15, 1: lanes 16-31
    const int m    = lane & 15;

    const long rowBlock = (long)blockIdx.x * 10 + wave;
    const long row      = rowBlock * 16 + m;

    v8f acc[NT];
#pragma unroll
    for (int t = 0; t < NT; ++t)
        acc[t] = (v8f){0.f, 0.f, 0.f, 0.f, 0.f, 0.f, 0.f, 0.f};

    const float*  xr = X + row * KIN + half * 8;
    const __bf16* wf = Wfrag + (size_t)lane * 16;

    for (int kc = 0; kc < KCHUNKS; ++kc) {
        if (kc + 1 < KCHUNKS) __builtin_prefetch(xr + 32, 0, 1);

        // A fragment: f32 -> bf16 in registers (16-bit A 16x32 layout)
        v16bf a;
#pragma unroll
        for (int j = 0; j < 8; ++j) {
            a[j]     = (__bf16)xr[j];
            a[8 + j] = (__bf16)xr[16 + j];
        }
        xr += 32;

#pragma unroll
        for (int t = 0; t < NT; ++t) {
            // one contiguous 32B per-lane load; coalesced 1KB per wave
            v16bf b = *(const v16bf*)(wf + ((size_t)(kc * NT + t) * 32) * 16);
            acc[t] = __builtin_amdgcn_wmma_f32_16x16x32_bf16(
                false, a, false, b, (short)0, acc[t], false, false);
        }
    }

    // D layout: VGPR r -> M = half*8 + r, N = m
    const int mbase = half * 8;
#pragma unroll
    for (int t = 0; t < NT; ++t) {
#pragma unroll
        for (int r = 0; r < 8; ++r) {
            C[(rowBlock * 16 + mbase + r) * (long)KOUT + t * 16 + m] = acc[t][r];
        }
    }
}

// --------------------------- edge aggregation ------------------------------
// agg[dst] += hw[src] * dinv[src]*dinv[dst]; one thread per (edge, float4 chunk)
template <int CH>   // CH = D/4 chunks per edge (power of two)
__global__ void edge_agg(const long long* __restrict__ ei,
                         const float* __restrict__ hw,
                         const float* __restrict__ dinv,
                         float* __restrict__ agg,
                         long long E) {
    constexpr int D  = CH * 4;
    constexpr int SH = (CH == 32) ? 5 : 2;
    long long tid = (long long)blockIdx.x * blockDim.x + threadIdx.x;
    if (tid >= E * CH) return;
    long long e = tid >> SH;
    int c = (int)(tid & (CH - 1));
    int s = (int)ei[e];
    int d = (int)ei[E + e];
    float norm = dinv[s] * dinv[d];
    const float4 v = *reinterpret_cast<const float4*>(hw + (size_t)s * D + c * 4);
    float* o = agg + (size_t)d * D + c * 4;
    __hip_atomic_fetch_add(o + 0, v.x * norm, __ATOMIC_RELAXED, __HIP_MEMORY_SCOPE_AGENT);
    __hip_atomic_fetch_add(o + 1, v.y * norm, __ATOMIC_RELAXED, __HIP_MEMORY_SCOPE_AGENT);
    __hip_atomic_fetch_add(o + 2, v.z * norm, __ATOMIC_RELAXED, __HIP_MEMORY_SCOPE_AGENT);
    __hip_atomic_fetch_add(o + 3, v.w * norm, __ATOMIC_RELAXED, __HIP_MEMORY_SCOPE_AGENT);
}

// ------------------------- finalize (self-loop+bias+relu) ------------------

__global__ void kFinalize128(float* __restrict__ agg, const float* __restrict__ hw,
                             const float* __restrict__ dinv,
                             const float* __restrict__ bias, int n) {
    int tid = blockIdx.x * blockDim.x + threadIdx.x;
    if (tid >= n * 128) return;
    int i    = tid >> 7;
    int dcol = tid & 127;
    float di = dinv[i];
    float v  = agg[tid] + hw[tid] * di * di + bias[dcol];
    agg[tid] = (v > 0.0f) ? v : 0.0f;
}

__global__ void kFinalizeOut(const float* __restrict__ agg, const float* __restrict__ hw,
                             const float* __restrict__ dinv,
                             const float* __restrict__ bias,
                             float* __restrict__ out, int n) {
    int tid = blockIdx.x * blockDim.x + threadIdx.x;
    if (tid >= n * 5) return;
    int i    = tid / 5;
    int dcol = tid - i * 5;
    float di = dinv[i];
    int idx  = i * 16 + dcol;          // padded layout
    out[tid] = agg[idx] + hw[idx] * di * di + bias[dcol];
}

// ------------------------------- launcher ----------------------------------

extern "C" void kernel_launch(void* const* d_in, const int* in_sizes, int n_in,
                              void* d_out, int out_size, void* d_ws, size_t ws_size,
                              hipStream_t stream) {
    (void)in_sizes; (void)n_in; (void)out_size; (void)ws_size;

    const float*     x  = (const float*)d_in[0];
    const long long* ei = (const long long*)d_in[1];   // int64 edge_index [2, E]
    const float*     W1 = (const float*)d_in[2];
    const float*     b1 = (const float*)d_in[3];
    const float*     W2 = (const float*)d_in[4];
    const float*     b2 = (const float*)d_in[5];
    const float*     W3 = (const float*)d_in[6];
    const float*     b3 = (const float*)d_in[7];
    float* out = (float*)d_out;

    const int  N = NNODES;
    const long long E = NEDGES;
    const int  H = 128, OUTD = 5, OPAD = 16;

    // workspace carve-up
    char* ws = (char*)d_ws;
    float* dinv = (float*)ws;
    size_t off = ((size_t)N * 4 + 511) & ~(size_t)511;
    float* bufA = (float*)(ws + off);                  // GEMM output (hw), 51.2 MB
    off += (size_t)N * H * 4;
    float* bufB = (float*)(ws + off);                  // agg / layer output, 51.2 MB
    off += (size_t)N * H * 4;
    __bf16* wf1 = (__bf16*)(ws + off);                 // 12*8*512 = 49152 bf16
    off += ((size_t)49152 * 2 + 255) & ~(size_t)255;
    __bf16* wf2 = (__bf16*)(ws + off);                 // 4*8*512  = 16384 bf16
    off += ((size_t)16384 * 2 + 255) & ~(size_t)255;
    __bf16* wf3 = (__bf16*)(ws + off);                 // 4*1*512  = 2048 bf16
    float* hw3  = bufA;                                // N x 16 (padded), inside bufA
    float* agg3 = bufA + (size_t)N * OPAD;             // N x 16, inside bufA

    // ---- normalization: dinv = rsqrt(deg), deg includes self-loops
    kInitDeg<<<(N + 255) / 256, 256, 0, stream>>>(dinv, N);
    kDegEdges<<<(unsigned)((E + 255) / 256), 256, 0, stream>>>(ei, dinv, E);
    kRsqrt<<<(N + 255) / 256, 256, 0, stream>>>(dinv, N);

    // ---- pack W1/W2/W3 into WMMA B-fragment layout (bf16)
    kPackW<8><<<(49152 + 255) / 256, 256, 0, stream>>>(W1, wf1, 384, H);
    kPackW<8><<<(16384 + 255) / 256, 256, 0, stream>>>(W2, wf2, H, H);
    kPackW<1><<<(2048  + 255) / 256, 256, 0, stream>>>(W3, wf3, H, OUTD);

    const unsigned aggGrid128 = (unsigned)((E * 32 + 255) / 256);
    const unsigned aggGrid16  = (unsigned)((E * 4 + 255) / 256);
    const unsigned finGrid128 = (unsigned)(((long long)N * 128 + 255) / 256);

    // ---- layer 1: x(384) -> h1(128), ReLU
    gemm_wmma_bf16<8, 384><<<625, 320, 0, stream>>>(x, wf1, bufA);
    hipMemsetAsync(bufB, 0, (size_t)N * H * 4, stream);
    edge_agg<32><<<aggGrid128, 256, 0, stream>>>(ei, bufA, dinv, bufB, E);
    kFinalize128<<<finGrid128, 256, 0, stream>>>(bufB, bufA, dinv, b1, N);

    // ---- layer 2: h1(128) -> h2(128), ReLU   (bufB reused as agg after GEMM)
    gemm_wmma_bf16<8, 128><<<625, 320, 0, stream>>>(bufB, wf2, bufA);
    hipMemsetAsync(bufB, 0, (size_t)N * H * 4, stream);
    edge_agg<32><<<aggGrid128, 256, 0, stream>>>(ei, bufA, dinv, bufB, E);
    kFinalize128<<<finGrid128, 256, 0, stream>>>(bufB, bufA, dinv, b2, N);

    // ---- layer 3: h2(128) -> logits(5), padded to 16 cols
    gemm_wmma_bf16<1, 128><<<625, 320, 0, stream>>>(bufB, wf3, hw3);
    hipMemsetAsync(agg3, 0, (size_t)N * OPAD * 4, stream);
    edge_agg<4><<<aggGrid16, 256, 0, stream>>>(ei, hw3, dinv, agg3, E);
    kFinalizeOut<<<(unsigned)(((long long)N * OUTD + 255) / 256), 256, 0, stream>>>(
        agg3, hw3, dinv, b3, out, N);
}
